// NaiveRNN_27839978013422
// MI455X (gfx1250) — compile-verified, matrix operand order checked by reading, not run on hardware
//
#include <hip/hip_runtime.h>
#include <hip/hip_bf16.h>
#include <math.h>

// Problem dims (fixed by the reference)
#define B_ 64
#define S_ 2048
#define I_ 512
#define H_ 512
#define NBLK_RECUR 16

typedef __attribute__((ext_vector_type(16))) __bf16 v16bf;
typedef __attribute__((ext_vector_type(8)))  __bf16 v8bf;
typedef __attribute__((ext_vector_type(8)))  float  v8f;

// ---------------------------------------------------------------------------
// Prep kernels: fp32 -> bf16 weight conversion, bias fusion, barrier reset
// ---------------------------------------------------------------------------
__global__ void rnn_cvt_bf16(const float* __restrict__ src, __bf16* __restrict__ dst, int n) {
    int i = blockIdx.x * 256 + threadIdx.x;
    if (i < n) dst[i] = (__bf16)src[i];
}

__global__ void rnn_bias_sum(const float* __restrict__ bih, const float* __restrict__ bhh,
                             float* __restrict__ bias, unsigned* __restrict__ sync) {
    int i = blockIdx.x * 256 + threadIdx.x;
    if (i < H_) bias[i] = bih[i] + bhh[i];
    if (i == 0) { sync[0] = 0u; sync[1] = 0u; }   // grid-barrier counter + generation
}

// ---------------------------------------------------------------------------
// Phase 1: x_proj[s*B+b, h] = sum_i x[b,s,i]*w_ih[h,i] + bias[h]
// One wave per 16x16 output tile; K looped in chunks of 32 (bf16 WMMA).
// ---------------------------------------------------------------------------
__global__ __launch_bounds__(256) void rnn_xproj(const float* __restrict__ x,     // [B,S,I]
                                                 const __bf16* __restrict__ wih,  // [H,I] bf16
                                                 const float* __restrict__ bias,  // [H]
                                                 float* __restrict__ out)         // [S*B,H]
{
    const int wave  = threadIdx.x >> 5;
    const int lane  = threadIdx.x & 31;
    const int tile  = blockIdx.x * 8 + wave;   // 262144 tiles total
    const int ntile = tile & 31;               // H/16 = 32 column tiles
    const int mtile = tile >> 5;               // (B*S)/16 = 8192 row tiles
    const int n15   = lane & 15;
    const int g     = lane >> 4;

    const int row = mtile * 16 + n15;          // output row r = s*B + b
    const int b   = row & (B_ - 1);
    const int s   = row >> 6;                  // B_ == 64
    const float*  xrow = x + ((size_t)b * S_ + s) * I_;
    const int     col  = ntile * 16 + n15;
    const __bf16* wrow = wih + (size_t)col * I_;

    const float bv = bias[col];
    v8f acc;
#pragma unroll
    for (int r = 0; r < 8; ++r) acc[r] = bv;

    for (int kc = 0; kc < I_; kc += 32) {
        v16bf a;
        const float* pa0 = xrow + kc + 8 * g;
        const float* pa1 = pa0 + 16;
#pragma unroll
        for (int j = 0; j < 8; ++j) { a[j] = (__bf16)pa0[j]; a[8 + j] = (__bf16)pa1[j]; }
        v16bf bm;
        v8bf b0 = *(const v8bf*)(wrow + kc + 8 * g);
        v8bf b1 = *(const v8bf*)(wrow + kc + 16 + 8 * g);
#pragma unroll
        for (int j = 0; j < 8; ++j) { bm[j] = b0[j]; bm[8 + j] = b1[j]; }

        acc = __builtin_amdgcn_wmma_f32_16x16x32_bf16(false, a, false, bm,
                                                      (short)0, acc, false, false);
    }

    float* o = out + ((size_t)(mtile * 16 + g * 8)) * H_ + col;
#pragma unroll
    for (int r = 0; r < 8; ++r) o[(size_t)r * H_] = acc[r];
}

// ---------------------------------------------------------------------------
// Phase 2: persistent recurrence kernel. 16 blocks x 8 waves = 128 waves,
// one 16x16 tile each (4 m-tiles x 32 n-tiles). Each wave keeps its W_hh
// B-operand slice (16 x v16bf = 128 VGPRs) resident in registers for all
// 2048 steps; steps are separated by an L2-atomic grid barrier instead of
// kernel launches.
// ---------------------------------------------------------------------------
__global__ __launch_bounds__(256) void rnn_recur(const __bf16* __restrict__ whh,  // [H,H] bf16
                                                 __bf16* __restrict__ h0,         // [B,H] bf16
                                                 __bf16* __restrict__ h1,         // [B,H] bf16
                                                 float* __restrict__ seq,         // [S*B,H]
                                                 unsigned* __restrict__ sync)     // [2]
{
    const int wave  = threadIdx.x >> 5;
    const int lane  = threadIdx.x & 31;
    const int tile  = blockIdx.x * 8 + wave;   // 0..127
    const int ntile = tile & 31;
    const int mtile = tile >> 5;               // 0..3
    const int n15   = lane & 15;
    const int g     = lane >> 4;

    const int mrow = mtile * 16 + n15;         // batch index b
    const int col  = ntile * 16 + n15;
    const __bf16* wrow = whh + (size_t)col * H_;

    // Hoist the wave's entire W_hh slice into VGPRs (loaded once, used 2048x)
    v16bf wreg[16];
#pragma unroll
    for (int c = 0; c < 16; ++c) {
        v8bf b0 = *(const v8bf*)(wrow + c * 32 + 8 * g);
        v8bf b1 = *(const v8bf*)(wrow + c * 32 + 16 + 8 * g);
#pragma unroll
        for (int j = 0; j < 8; ++j) { wreg[c][j] = b0[j]; wreg[c][8 + j] = b1[j]; }
    }

    for (int s = 0; s < S_; ++s) {
        const __bf16* hp = (s & 1) ? h1 : h0;
        __bf16*       hn = (s & 1) ? h0 : h1;
        const __bf16* arow = hp + (size_t)mrow * H_;

        v8f acc = {};
#pragma unroll
        for (int c = 0; c < 16; ++c) {
            v16bf a;
            v8bf a0 = *(const v8bf*)(arow + c * 32 + 8 * g);
            v8bf a1 = *(const v8bf*)(arow + c * 32 + 16 + 8 * g);
#pragma unroll
            for (int j = 0; j < 8; ++j) { a[j] = a0[j]; a[8 + j] = a1[j]; }
            acc = __builtin_amdgcn_wmma_f32_16x16x32_bf16(false, a, false, wreg[c],
                                                          (short)0, acc, false, false);
        }

        const int m0 = mtile * 16 + g * 8;
        float*  srow = seq + ((size_t)s * B_ + m0) * H_ + col;
        __bf16* hrow = hn + (size_t)m0 * H_ + col;
#pragma unroll
        for (int r = 0; r < 8; ++r) {
            float hv = tanhf(srow[(size_t)r * H_] + acc[r]);
            srow[(size_t)r * H_] = hv;             // hidden_seq output (fp32)
            hrow[(size_t)r * H_] = (__bf16)hv;     // next-step A operand (bf16)
        }

        // --- grid-wide barrier: publish hn, wait for all 16 blocks ---
        __syncthreads();
        if (threadIdx.x == 0) {
            __threadfence();                       // release this block's h writes
            unsigned gen = __hip_atomic_load(&sync[1], __ATOMIC_RELAXED,
                                             __HIP_MEMORY_SCOPE_AGENT);
            unsigned old = __hip_atomic_fetch_add(&sync[0], 1u, __ATOMIC_ACQ_REL,
                                                  __HIP_MEMORY_SCOPE_AGENT);
            if (old == (unsigned)(NBLK_RECUR - 1)) {
                __hip_atomic_store(&sync[0], 0u, __ATOMIC_RELAXED,
                                   __HIP_MEMORY_SCOPE_AGENT);
                __hip_atomic_store(&sync[1], gen + 1u, __ATOMIC_RELEASE,
                                   __HIP_MEMORY_SCOPE_AGENT);
            } else {
                while (__hip_atomic_load(&sync[1], __ATOMIC_ACQUIRE,
                                         __HIP_MEMORY_SCOPE_AGENT) == gen) {
                    __builtin_amdgcn_s_sleep(1);
                }
            }
        }
        __syncthreads();
    }
}

// ---------------------------------------------------------------------------
// Final output: h_last.T [H, B] from step S-1 rows of hidden_seq.
// ---------------------------------------------------------------------------
__global__ void rnn_last_t(const float* __restrict__ seq, float* __restrict__ out2) {
    int i = blockIdx.x * 256 + threadIdx.x;    // H*B elements
    if (i < H_ * B_) {
        int h = i / B_;
        int b = i % B_;
        out2[i] = seq[((size_t)(S_ - 1) * B_ + b) * H_ + h];
    }
}

// ---------------------------------------------------------------------------
extern "C" void kernel_launch(void* const* d_in, const int* in_sizes, int n_in,
                              void* d_out, int out_size, void* d_ws, size_t ws_size,
                              hipStream_t stream) {
    const float* inputs = (const float*)d_in[0];  // [B,S,I]
    const float* state  = (const float*)d_in[1];  // [B,H]
    const float* w_ih   = (const float*)d_in[2];  // [H,I]
    const float* b_ih   = (const float*)d_in[3];  // [H,1]
    const float* w_hh   = (const float*)d_in[4];  // [H,H]
    const float* b_hh   = (const float*)d_in[5];  // [H,1]
    float* out = (float*)d_out;

    // Workspace layout (~1.2 MB, L2-resident):
    char* ws = (char*)d_ws;
    __bf16*   wih_bf = (__bf16*)(ws);                     // 512 KB
    __bf16*   whh_bf = (__bf16*)(ws + 524288);            // 512 KB
    float*    bias   = (float*)(ws + 1048576);            // 2 KB
    __bf16*   h_buf0 = (__bf16*)(ws + 1050624);           // 64 KB
    __bf16*   h_buf1 = (__bf16*)(ws + 1116160);           // 64 KB
    unsigned* sync   = (unsigned*)(ws + 1181696);         // 8 B barrier state

    // Prep: weight/state conversion, bias fusion, barrier reset
    rnn_cvt_bf16<<<(H_ * I_ + 255) / 256, 256, 0, stream>>>(w_ih, wih_bf, H_ * I_);
    rnn_cvt_bf16<<<(H_ * H_ + 255) / 256, 256, 0, stream>>>(w_hh, whh_bf, H_ * H_);
    rnn_cvt_bf16<<<(B_ * H_ + 255) / 256, 256, 0, stream>>>(state, h_buf0, B_ * H_);
    rnn_bias_sum<<<2, 256, 0, stream>>>(b_ih, b_hh, bias, sync);

    // Phase 1: input projection GEMM straight into hidden_seq region of d_out
    rnn_xproj<<<(8192 * 32) / 8, 256, 0, stream>>>(inputs, wih_bf, bias, out);

    // Phase 2: single persistent kernel runs all 2048 steps with device-side
    // grid barriers (no per-step launch overhead)
    rnn_recur<<<NBLK_RECUR, 256, 0, stream>>>(whh_bf, h_buf0, h_buf1, out, sync);

    // h_last.T appended after hidden_seq
    rnn_last_t<<<(H_ * B_ + 255) / 256, 256, 0, stream>>>(out, out + (size_t)S_ * B_ * H_);
}
